// FPNIoUNetHR_85023172592430
// MI455X (gfx1250) — matrix operand order repaired and checked by reading, not compiled, not run on hardware
//
#include <hip/hip_runtime.h>
#include <hip/hip_bf16.h>

// ---------------------------------------------------------------------------
// MI455X (gfx1250) implementation of the IoU-Net head.
//
// Roofline: ~3e11 FLOPs dominated by 3x conv3x3(256->256, 64x36x36) and the
// [1024x12544]x[12544x1024] linear. HBM traffic is only a few hundred MB
// (~15us at 23.3 TB/s), so the GEMMs must run on the matrix pipe:
// v_wmma_f32_16x16x32_f16 (f16 in, f32 accum). Convs are implicit GEMM over
// NHWC f16 activations with 128x128 block tiles, 32x64 per-wave tiles and
// double-buffered LDS (one barrier per K-step, global loads issued before the
// WMMAs so HBM latency hides behind matrix math). BN is a deterministic
// 2-stage reduction; PrRoIPool uses the exact separable hat-integral form.
// ---------------------------------------------------------------------------

typedef __attribute__((ext_vector_type(16))) _Float16 v16h;
typedef __attribute__((ext_vector_type(8)))  _Float16 v8h;
typedef __attribute__((ext_vector_type(8)))  float    v8f;

#define LDSW 40   // halves per LDS tile row: 32 data + 8 pad (80B, 16B aligned)

// --- WMMA fragment loaders (per ISA 7.12.2 16-bit layouts) ------------------
// A 16x32: lanes 0-15 -> M=lane, K = {0..7,16..23}; lanes 16-31 -> {8..15,24..31}
__device__ __forceinline__ v16h load_frag_a(const _Float16* lds, int row, int half) {
  const _Float16* p = lds + row * LDSW + (half ? 8 : 0);
  v8h lo = *(const v8h*)p;
  v8h hi = *(const v8h*)(p + 16);
  return __builtin_shufflevector(lo, hi, 0,1,2,3,4,5,6,7,8,9,10,11,12,13,14,15);
}
// B 32x16 stored n-major in LDS: lane holds col n=lane&15, K={0..7,16..23} (lanes 0-15)
// or K={8..15,24..31} (lanes 16-31).
__device__ __forceinline__ v16h load_frag_b(const _Float16* lds, int row, int half) {
  const _Float16* p = lds + row * LDSW + (half ? 8 : 0);
  v8h lo = *(const v8h*)p;
  v8h hi = *(const v8h*)(p + 16);
  return __builtin_shufflevector(lo, hi, 0,1,2,3,4,5,6,7,8,9,10,11,12,13,14,15);
}

#define WMMA_F16(A, B, C) \
  __builtin_amdgcn_wmma_f32_16x16x32_f16(false, (A), false, (B), (short)0, (C), false, false)

// ===========================================================================
// 128x128 double-buffered GEMM: Y[M,N](f32) = A[M,K](f16 row-major) *
// Bn[N,K](f16 n-major) + bias.  256 thr = 8 waves; wave tile 32x64 (8 WMMA).
// Requires M%128==0, N%128==0, K%32==0.
// ===========================================================================
__global__ __launch_bounds__(256) void k_gemm128_wmma(
    const _Float16* __restrict__ A, const _Float16* __restrict__ B,
    const float* __restrict__ bias, float* __restrict__ Y, int N, int K)
{
  __shared__ __align__(16) _Float16 lA[2][128 * LDSW];
  __shared__ __align__(16) _Float16 lB[2][128 * LDSW];
  const int tid = threadIdx.x;
  const int m0 = blockIdx.y * 128, n0 = blockIdx.x * 128;
  const int row = tid >> 1, seg = tid & 1;          // 128 rows x 2 segments
  const int lane = tid & 31, wave = tid >> 5;
  const int wm = (wave & 3) * 32;                   // 4 m-waves
  const int wn = (wave >> 2) * 64;                  // 2 n-waves
  const int lm = lane & 15, half = lane >> 4;
  const int soff = row * LDSW + seg * 16;

  v8f acc[2][4];
#pragma unroll
  for (int i = 0; i < 2; ++i)
#pragma unroll
    for (int j = 0; j < 4; ++j)
      acc[i][j] = v8f{0.f,0.f,0.f,0.f,0.f,0.f,0.f,0.f};

  const _Float16* gA = A + (size_t)(m0 + row) * K + seg * 16;
  const _Float16* gB = B + (size_t)(n0 + row) * K + seg * 16;

  // preload tile kb=0 into buffer 0
  {
    uint4 a0 = *(const uint4*)(gA);
    uint4 a1 = *(const uint4*)(gA + 8);
    uint4 b0 = *(const uint4*)(gB);
    uint4 b1 = *(const uint4*)(gB + 8);
    *(uint4*)&lA[0][soff] = a0; *(uint4*)&lA[0][soff + 8] = a1;
    *(uint4*)&lB[0][soff] = b0; *(uint4*)&lB[0][soff + 8] = b1;
  }
  __syncthreads();

  int cur = 0;
  for (int kb = 0; kb < K; kb += 32) {
    const int nxt = kb + 32;
    const bool has = nxt < K;
    uint4 na0, na1, nb0, nb1;
    if (has) {                                   // issue next-tile loads first
      na0 = *(const uint4*)(gA + nxt);
      na1 = *(const uint4*)(gA + nxt + 8);
      nb0 = *(const uint4*)(gB + nxt);
      nb1 = *(const uint4*)(gB + nxt + 8);
      if (nxt + 32 < K) {                        // global_prefetch_b8
        __builtin_prefetch(gA + nxt + 32, 0, 1);
        __builtin_prefetch(gB + nxt + 32, 0, 1);
      }
    }
    // compute on current buffer while loads are in flight
    v16h fa[2], fb[4];
#pragma unroll
    for (int i = 0; i < 2; ++i) fa[i] = load_frag_a(lA[cur], wm + i * 16 + lm, half);
#pragma unroll
    for (int j = 0; j < 4; ++j) fb[j] = load_frag_b(lB[cur], wn + j * 16 + lm, half);
#pragma unroll
    for (int i = 0; i < 2; ++i)
#pragma unroll
      for (int j = 0; j < 4; ++j)
        acc[i][j] = WMMA_F16(fa[i], fb[j], acc[i][j]);

    if (has) {
      _Float16* dA = &lA[cur ^ 1][soff];
      _Float16* dB = &lB[cur ^ 1][soff];
      *(uint4*)dA = na0; *(uint4*)(dA + 8) = na1;
      *(uint4*)dB = nb0; *(uint4*)(dB + 8) = nb1;
    }
    __syncthreads();
    cur ^= 1;
  }
#pragma unroll
  for (int i = 0; i < 2; ++i)
#pragma unroll
    for (int j = 0; j < 4; ++j) {
      int n = n0 + wn + j * 16 + lm;
      int mb = m0 + wm + i * 16 + half * 8;
      float bv = bias[n];
#pragma unroll
      for (int r = 0; r < 8; ++r)
        Y[(size_t)(mb + r) * N + n] = acc[i][j][r] + bv;
    }
}

// ===========================================================================
// Implicit-GEMM 3x3 conv, pad 1, NHWC f16 (S=64, H=W=36, Ci=Co=256).
// Same 128x128 double-buffered scheme; M = 64*36*36 = 82944 (648 tiles).
// ===========================================================================
__global__ __launch_bounds__(256) void k_conv3x3_wmma(
    const _Float16* __restrict__ F, const _Float16* __restrict__ Bw,
    const float* __restrict__ bias, float* __restrict__ Y)
{
  const int K = 2304, N = 256;
  __shared__ __align__(16) _Float16 lA[2][128 * LDSW];
  __shared__ __align__(16) _Float16 lB[2][128 * LDSW];
  const int tid = threadIdx.x;
  const int m0 = blockIdx.y * 128, n0 = blockIdx.x * 128;
  const int row = tid >> 1, seg = tid & 1;
  const int lane = tid & 31, wave = tid >> 5;
  const int wm = (wave & 3) * 32;
  const int wn = (wave >> 2) * 64;
  const int lm = lane & 15, half = lane >> 4;
  const int soff = row * LDSW + seg * 16;

  const int p = m0 + row;
  const int s = p / 1296, hw = p - s * 1296;
  const int h = hw / 36, w = hw - h * 36;

  v8f acc[2][4];
#pragma unroll
  for (int i = 0; i < 2; ++i)
#pragma unroll
    for (int j = 0; j < 4; ++j)
      acc[i][j] = v8f{0.f,0.f,0.f,0.f,0.f,0.f,0.f,0.f};

  const _Float16* gB = Bw + (size_t)(n0 + row) * K + seg * 16;

  auto loadA = [&](int kb, uint4& x0, uint4& x1) {
    const int khw = kb >> 8, c0 = kb & 255;
    const int hh = h + khw / 3 - 1, ww = w + khw % 3 - 1;
    x0 = uint4{0,0,0,0}; x1 = uint4{0,0,0,0};
    if ((unsigned)hh < 36u && (unsigned)ww < 36u) {
      const _Float16* gp = F + ((size_t)((s * 36 + hh) * 36 + ww)) * 256 + c0 + seg * 16;
      x0 = *(const uint4*)gp;
      x1 = *(const uint4*)(gp + 8);
    }
  };

  {
    uint4 a0, a1;
    loadA(0, a0, a1);
    uint4 b0 = *(const uint4*)(gB);
    uint4 b1 = *(const uint4*)(gB + 8);
    *(uint4*)&lA[0][soff] = a0; *(uint4*)&lA[0][soff + 8] = a1;
    *(uint4*)&lB[0][soff] = b0; *(uint4*)&lB[0][soff + 8] = b1;
  }
  __syncthreads();

  int cur = 0;
  for (int kb = 0; kb < K; kb += 32) {
    const int nxt = kb + 32;
    const bool has = nxt < K;
    uint4 na0, na1, nb0, nb1;
    if (has) {
      loadA(nxt, na0, na1);
      nb0 = *(const uint4*)(gB + nxt);
      nb1 = *(const uint4*)(gB + nxt + 8);
      if (nxt + 32 < K) __builtin_prefetch(gB + nxt + 32, 0, 1);
    }
    v16h fa[2], fb[4];
#pragma unroll
    for (int i = 0; i < 2; ++i) fa[i] = load_frag_a(lA[cur], wm + i * 16 + lm, half);
#pragma unroll
    for (int j = 0; j < 4; ++j) fb[j] = load_frag_b(lB[cur], wn + j * 16 + lm, half);
#pragma unroll
    for (int i = 0; i < 2; ++i)
#pragma unroll
      for (int j = 0; j < 4; ++j)
        acc[i][j] = WMMA_F16(fa[i], fb[j], acc[i][j]);

    if (has) {
      _Float16* dA = &lA[cur ^ 1][soff];
      _Float16* dB = &lB[cur ^ 1][soff];
      *(uint4*)dA = na0; *(uint4*)(dA + 8) = na1;
      *(uint4*)dB = nb0; *(uint4*)(dB + 8) = nb1;
    }
    __syncthreads();
    cur ^= 1;
  }
#pragma unroll
  for (int i = 0; i < 2; ++i)
#pragma unroll
    for (int j = 0; j < 4; ++j) {
      int n = n0 + wn + j * 16 + lm;
      int mb = m0 + wm + i * 16 + half * 8;
      float bv = bias[n];
#pragma unroll
      for (int r = 0; r < 8; ++r)
        Y[(size_t)(mb + r) * N + n] = acc[i][j][r] + bv;
    }
}

// ===========================================================================
// 64x64 GEMM for the small-M (M=64) layers: f1r, f2r.
// ===========================================================================
__global__ __launch_bounds__(128) void k_gemm_wmma(
    const _Float16* __restrict__ A, const _Float16* __restrict__ B,
    const float* __restrict__ bias, float* __restrict__ Y, int N, int K)
{
  __shared__ __align__(16) _Float16 lA[64 * LDSW];
  __shared__ __align__(16) _Float16 lB[64 * LDSW];
  const int tid = threadIdx.x;
  const int m0 = blockIdx.y * 64, n0 = blockIdx.x * 64;
  const int row = tid >> 1, seg = tid & 1;
  const int lane = tid & 31, wave = tid >> 5;
  const int wm = (wave & 1) * 32, wn = (wave >> 1) * 32;
  const int lm = lane & 15, half = lane >> 4;

  v8f acc[2][2];
#pragma unroll
  for (int i = 0; i < 2; ++i)
#pragma unroll
    for (int j = 0; j < 2; ++j)
      acc[i][j] = v8f{0.f,0.f,0.f,0.f,0.f,0.f,0.f,0.f};

  const _Float16* gA = A + (size_t)(m0 + row) * K + seg * 16;
  const _Float16* gB = B + (size_t)(n0 + row) * K + seg * 16;
  _Float16* sA = &lA[row * LDSW + seg * 16];
  _Float16* sB = &lB[row * LDSW + seg * 16];

  for (int kb = 0; kb < K; kb += 32) {
    uint4 a0 = *(const uint4*)(gA + kb);
    uint4 a1 = *(const uint4*)(gA + kb + 8);
    uint4 b0 = *(const uint4*)(gB + kb);
    uint4 b1 = *(const uint4*)(gB + kb + 8);
    if (kb + 32 < K) {
      __builtin_prefetch(gA + kb + 32, 0, 1);
      __builtin_prefetch(gB + kb + 32, 0, 1);
    }
    *(uint4*)sA = a0; *(uint4*)(sA + 8) = a1;
    *(uint4*)sB = b0; *(uint4*)(sB + 8) = b1;
    __syncthreads();
    v16h fa0 = load_frag_a(lA, wm + lm,      half);
    v16h fa1 = load_frag_a(lA, wm + 16 + lm, half);
    v16h fb0 = load_frag_b(lB, wn + lm,      half);
    v16h fb1 = load_frag_b(lB, wn + 16 + lm, half);
    acc[0][0] = WMMA_F16(fa0, fb0, acc[0][0]);
    acc[0][1] = WMMA_F16(fa0, fb1, acc[0][1]);
    acc[1][0] = WMMA_F16(fa1, fb0, acc[1][0]);
    acc[1][1] = WMMA_F16(fa1, fb1, acc[1][1]);
    __syncthreads();
  }
#pragma unroll
  for (int i = 0; i < 2; ++i)
#pragma unroll
    for (int j = 0; j < 2; ++j) {
      int n = n0 + wn + j * 16 + lm;
      int mb = m0 + wm + i * 16 + half * 8;
      float bv = bias[n];
#pragma unroll
      for (int r = 0; r < 8; ++r)
        Y[(size_t)(mb + r) * N + n] = acc[i][j][r] + bv;
    }
}

// --- BatchNorm (training stats), deterministic two-stage --------------------
__global__ __launch_bounds__(256) void k_bn_stats(
    const float* __restrict__ y, float* __restrict__ part,
    int R, int C, int rowsPerWG)
{
  const int wg = blockIdx.x, tid = threadIdx.x;
  int r0 = wg * rowsPerWG, r1 = r0 + rowsPerWG;
  if (r1 > R) r1 = R;
  const int nch = C >> 8;  // C/256 (C is 256 or 1024)
  float s[4] = {0.f,0.f,0.f,0.f}, q[4] = {0.f,0.f,0.f,0.f};
  for (int r = r0; r < r1; ++r) {
    const float* yr = y + (size_t)r * C + tid;
#pragma unroll
    for (int j = 0; j < 4; ++j)
      if (j < nch) { float v = yr[j * 256]; s[j] += v; q[j] += v * v; }
  }
#pragma unroll
  for (int j = 0; j < 4; ++j)
    if (j < nch) {
      part[((size_t)wg * 2 + 0) * C + j * 256 + tid] = s[j];
      part[((size_t)wg * 2 + 1) * C + j * 256 + tid] = q[j];
    }
}

__global__ __launch_bounds__(256) void k_bn_finalize(
    const float* __restrict__ part, const float* __restrict__ g,
    const float* __restrict__ be, float* __restrict__ ab, int P, int C, float Rf)
{
  for (int c = threadIdx.x; c < C; c += 256) {
    float s = 0.f, q = 0.f;
    for (int p = 0; p < P; ++p) {
      s += part[((size_t)p * 2 + 0) * C + c];
      q += part[((size_t)p * 2 + 1) * C + c];
    }
    float mean = s / Rf;
    float var = q / Rf - mean * mean;
    if (var < 0.f) var = 0.f;
    float a = g[c] * rsqrtf(var + 1e-5f);
    ab[c] = a;
    ab[C + c] = be[c] - mean * a;
  }
}

__global__ __launch_bounds__(256) void k_bn_apply(
    const float* __restrict__ y, const float* __restrict__ ab,
    _Float16* __restrict__ oh, float* __restrict__ of,
    int cmask, int C, long total)
{
  long i = (long)blockIdx.x * 256 + threadIdx.x;
  if (i >= total) return;
  int c = (int)(i & cmask);
  float v = y[i] * ab[c] + ab[C + c];
  v = fmaxf(v, 0.f);
  oh[i] = (_Float16)v;
  if (of) of[i] = v;
}

// --- layout / conversion kernels -------------------------------------------
__global__ __launch_bounds__(256) void k_cvt_f16(
    const float* __restrict__ in, _Float16* __restrict__ out, long total)
{
  long i = (long)blockIdx.x * 256 + threadIdx.x;
  if (i < total) out[i] = (_Float16)in[i];
}

__global__ __launch_bounds__(256) void k_pack_conv(
    const float* __restrict__ w, _Float16* __restrict__ bp, int total)
{
  int i = blockIdx.x * 256 + threadIdx.x;
  if (i >= total) return;
  int kk = i % 9;
  int ci = (i / 9) & 255;
  int o = i / (9 * 256);
  bp[(size_t)o * 2304 + kk * 256 + ci] = (_Float16)w[i];
}

__global__ __launch_bounds__(256) void k_nchw_to_nhwc(
    const float* __restrict__ in, _Float16* __restrict__ out, long total)
{
  long i = (long)blockIdx.x * 256 + threadIdx.x;
  if (i >= total) return;
  long hw = i % 1296;
  long sc = i / 1296;
  int c = (int)(sc & 255);
  long s = sc >> 8;
  out[(s * 1296 + hw) * 256 + c] = (_Float16)in[i];
}

// --- exact PrRoIPool (separable hat-integral weights) ------------------------
__device__ __forceinline__ float hat_int(float x) {
  float u = fminf(1.f, fmaxf(-1.f, x));
  return (u <= 0.f) ? 0.5f * (1.f + u) * (1.f + u) : (0.5f + u - 0.5f * u * u);
}

// pooled_r[s, c*25 + py*5 + px]  (feat NHWC f16)
__global__ __launch_bounds__(256) void k_pool_r(
    const _Float16* __restrict__ feat, const float* __restrict__ bb1,
    _Float16* __restrict__ out)
{
  __shared__ float Wx[180], Wy[180];
  const int s = blockIdx.x;
  const float* r = bb1 + (size_t)s * 4;      // [1,S,4] xywh
  float x1 = r[0] * 0.125f, y1 = r[1] * 0.125f;
  float x2 = (r[0] + r[2]) * 0.125f, y2 = (r[1] + r[3]) * 0.125f;
  float bw = (x2 - x1) * 0.2f, bh = (y2 - y1) * 0.2f;
  for (int i = threadIdx.x; i < 360; i += 256) {
    if (i < 180) {
      int px = i / 36, w = i % 36;
      float xs = x1 + px * bw;
      Wx[i] = hat_int(xs + bw - (float)w) - hat_int(xs - (float)w);
    } else {
      int j = i - 180, py = j / 36, h = j % 36;
      float ys = y1 + py * bh;
      Wy[j] = hat_int(ys + bh - (float)h) - hat_int(ys - (float)h);
    }
  }
  __syncthreads();
  const int c = threadIdx.x;  // 256 channels
  float acc[25];
#pragma unroll
  for (int k = 0; k < 25; ++k) acc[k] = 0.f;
  for (int h = 0; h < 36; ++h) {
    float ra[5] = {0.f,0.f,0.f,0.f,0.f};
    const _Float16* fp = feat + ((size_t)(s * 36 + h) * 36) * 256 + c;
    for (int w = 0; w < 36; ++w) {
      float v = (float)fp[(size_t)w * 256];
#pragma unroll
      for (int px = 0; px < 5; ++px) ra[px] += Wx[px * 36 + w] * v;
    }
#pragma unroll
    for (int py = 0; py < 5; ++py) {
      float wy = Wy[py * 36 + h];
#pragma unroll
      for (int px = 0; px < 5; ++px) acc[py * 5 + px] += wy * ra[px];
    }
  }
  float area = bw * bh;
  float sc = (area > 0.f) ? (1.f / fmaxf(area, 1e-12f)) : 0.f;
  _Float16* op = out + (size_t)s * 6400 + c * 25;
#pragma unroll
  for (int k = 0; k < 25; ++k) op[k] = (_Float16)(acc[k] * sc);
}

// pooled_t[(s*16+n), c*49 + py*7 + px], modulation folded in (linear pooling)
__global__ __launch_bounds__(256) void k_pool_t(
    const _Float16* __restrict__ feat, const float* __restrict__ props,
    const float* __restrict__ mod, _Float16* __restrict__ out)
{
  __shared__ float Wx[252], Wy[252];
  const int b = blockIdx.x;           // s*16+n
  const int s = b >> 4;
  const float* r = props + (size_t)b * 4;    // [1,S,N,4] xywh
  float x1 = r[0] * 0.125f, y1 = r[1] * 0.125f;
  float x2 = (r[0] + r[2]) * 0.125f, y2 = (r[1] + r[3]) * 0.125f;
  float bw = (x2 - x1) / 7.f, bh = (y2 - y1) / 7.f;
  for (int i = threadIdx.x; i < 504; i += 256) {
    if (i < 252) {
      int px = i / 36, w = i % 36;
      float xs = x1 + px * bw;
      Wx[i] = hat_int(xs + bw - (float)w) - hat_int(xs - (float)w);
    } else {
      int j = i - 252, py = j / 36, h = j % 36;
      float ys = y1 + py * bh;
      Wy[j] = hat_int(ys + bh - (float)h) - hat_int(ys - (float)h);
    }
  }
  __syncthreads();
  const int c = threadIdx.x;
  float acc[49];
#pragma unroll
  for (int k = 0; k < 49; ++k) acc[k] = 0.f;
  for (int h = 0; h < 36; ++h) {
    float ra[7] = {0.f,0.f,0.f,0.f,0.f,0.f,0.f};
    const _Float16* fp = feat + ((size_t)(s * 36 + h) * 36) * 256 + c;
    for (int w = 0; w < 36; ++w) {
      float v = (float)fp[(size_t)w * 256];
#pragma unroll
      for (int px = 0; px < 7; ++px) ra[px] += Wx[px * 36 + w] * v;
    }
#pragma unroll
    for (int py = 0; py < 7; ++py) {
      float wy = Wy[py * 36 + h];
#pragma unroll
      for (int px = 0; px < 7; ++px) acc[py * 7 + px] += wy * ra[px];
    }
  }
  float area = bw * bh;
  float sc = (area > 0.f) ? (1.f / fmaxf(area, 1e-12f)) : 0.f;
  sc *= mod[s * 256 + c];
  _Float16* op = out + (size_t)b * 12544 + c * 49;
#pragma unroll
  for (int k = 0; k < 49; ++k) op[k] = (_Float16)(acc[k] * sc);
}

// --- final IoU head: out[row] = dot(x[row,:1024], w_iou) + b_iou ------------
__global__ __launch_bounds__(256) void k_iou(
    const _Float16* __restrict__ x, const float* __restrict__ wv,
    const float* __restrict__ bv, float* __restrict__ out)
{
  __shared__ float red[256];
  const int row = blockIdx.x;
  float s = 0.f;
  for (int k = threadIdx.x; k < 1024; k += 256)
    s += (float)x[(size_t)row * 1024 + k] * wv[k];
  red[threadIdx.x] = s;
  __syncthreads();
  for (int o = 128; o > 0; o >>= 1) {
    if (threadIdx.x < (unsigned)o) red[threadIdx.x] += red[threadIdx.x + o];
    __syncthreads();
  }
  if (threadIdx.x == 0) out[row] = red[0] + bv[0];
}

// ---------------------------------------------------------------------------
extern "C" void kernel_launch(void* const* d_in, const int* in_sizes, int n_in,
                              void* d_out, int out_size, void* d_ws, size_t ws_size,
                              hipStream_t stream) {
  (void)in_sizes; (void)n_in; (void)out_size; (void)ws_size;
  const float* feat1  = (const float*)d_in[0];
  const float* feat2  = (const float*)d_in[1];
  const float* bb1    = (const float*)d_in[2];
  const float* props  = (const float*)d_in[3];
  const float* w_c1r  = (const float*)d_in[4];
  const float* b_c1r  = (const float*)d_in[5];
  const float* g_c1r  = (const float*)d_in[6];
  const float* be_c1r = (const float*)d_in[7];
  const float* w_c1t  = (const float*)d_in[8];
  const float* b_c1t  = (const float*)d_in[9];
  const float* g_c1t  = (const float*)d_in[10];
  const float* be_c1t = (const float*)d_in[11];
  const float* w_c2t  = (const float*)d_in[12];
  const float* b_c2t  = (const float*)d_in[13];
  const float* g_c2t  = (const float*)d_in[14];
  const float* be_c2t = (const float*)d_in[15];
  const float* w_f1r  = (const float*)d_in[16];
  const float* b_f1r  = (const float*)d_in[17];
  const float* g_f1r  = (const float*)d_in[18];
  const float* be_f1r = (const float*)d_in[19];
  const float* w_f2r  = (const float*)d_in[20];
  const float* b_f2r  = (const float*)d_in[21];
  const float* g_f2r  = (const float*)d_in[22];
  const float* be_f2r = (const float*)d_in[23];
  const float* w_f1rt  = (const float*)d_in[24];
  const float* b_f1rt  = (const float*)d_in[25];
  const float* g_f1rt  = (const float*)d_in[26];
  const float* be_f1rt = (const float*)d_in[27];
  const float* w_f2rt  = (const float*)d_in[28];
  const float* b_f2rt  = (const float*)d_in[29];
  const float* g_f2rt  = (const float*)d_in[30];
  const float* be_f2rt = (const float*)d_in[31];
  const float* w_iou   = (const float*)d_in[32];
  const float* b_iou   = (const float*)d_in[33];
  float* out = (float*)d_out;

  const long FEAT = 64L * 1296 * 256;    // 21,233,664
  char* base = (char*)d_ws;
  size_t off = 0;
  auto alloc = [&](size_t bytes) -> void* {
    void* p = base + off;
    off += (bytes + 255) & ~(size_t)255;
    return p;
  };
  _Float16* bufA = (_Float16*)alloc(FEAT * 2);        // feat NHWC f16 (reused)
  _Float16* bufB = (_Float16*)alloc(FEAT * 2);        // c1t f16
  _Float16* bufC = (_Float16*)alloc(FEAT * 2);        // c3_r f16
  float*    ybuf = (float*)   alloc(FEAT * 4);        // conv pre-BN scratch
  _Float16* w1r  = (_Float16*)alloc(2304L * 256 * 2);
  _Float16* w1t  = (_Float16*)alloc(2304L * 256 * 2);
  _Float16* w2t  = (_Float16*)alloc(2304L * 256 * 2);
  _Float16* wf1r = (_Float16*)alloc(1024L * 6400 * 2);
  _Float16* wf2r = (_Float16*)alloc(256L * 1024 * 2);
  _Float16* wf1rt= (_Float16*)alloc(1024L * 12544 * 2);
  _Float16* wf2rt= (_Float16*)alloc(1024L * 1024 * 2);
  _Float16* poolr= (_Float16*)alloc(64L * 6400 * 2);
  float*    y1   = (float*)   alloc(64L * 1024 * 4);
  _Float16* m_h  = (_Float16*)alloc(64L * 1024 * 2);
  float*    y2   = (float*)   alloc(64L * 256 * 4);
  float*    modf = (float*)   alloc(64L * 256 * 4);
  _Float16* modh = (_Float16*)alloc(64L * 256 * 2);
  _Float16* poolt= (_Float16*)alloc(1024L * 12544 * 2);
  float*    z1   = (float*)   alloc(1024L * 1024 * 4);
  _Float16* z1h  = (_Float16*)alloc(1024L * 1024 * 2);
  float*    z2   = (float*)   alloc(1024L * 1024 * 4);
  _Float16* z2h  = (_Float16*)alloc(1024L * 1024 * 2);
  float*    part = (float*)   alloc(1 << 20);         // BN partials
  float*    ab   = (float*)   alloc(2L * 1024 * 4);   // BN scale/shift

  auto blocks = [](long n) { return (unsigned)((n + 255) / 256); };

  // --- weight packing (f32 -> f16, n-major K layout) ---
  k_pack_conv<<<blocks(589824), 256, 0, stream>>>(w_c1r, w1r, 589824);
  k_pack_conv<<<blocks(589824), 256, 0, stream>>>(w_c1t, w1t, 589824);
  k_pack_conv<<<blocks(589824), 256, 0, stream>>>(w_c2t, w2t, 589824);
  k_cvt_f16<<<blocks(6553600), 256, 0, stream>>>(w_f1r,  wf1r,  6553600L);
  k_cvt_f16<<<blocks(262144),  256, 0, stream>>>(w_f2r,  wf2r,  262144L);
  k_cvt_f16<<<blocks(12845056),256, 0, stream>>>(w_f1rt, wf1rt, 12845056L);
  k_cvt_f16<<<blocks(1048576), 256, 0, stream>>>(w_f2rt, wf2rt, 1048576L);

  const dim3 convGrid(2, 648);   // N/128, M/128 (M = 64*36*36)
  // --- reference branch: conv_bn_relu(feat1) -> c3_r ---
  k_nchw_to_nhwc<<<blocks(FEAT), 256, 0, stream>>>(feat1, bufA, FEAT);
  k_conv3x3_wmma<<<convGrid, 256, 0, stream>>>(bufA, w1r, b_c1r, ybuf);
  k_bn_stats<<<324, 256, 0, stream>>>(ybuf, part, 82944, 256, 256);
  k_bn_finalize<<<1, 256, 0, stream>>>(part, g_c1r, be_c1r, ab, 324, 256, 82944.f);
  k_bn_apply<<<blocks(FEAT), 256, 0, stream>>>(ybuf, ab, bufC, nullptr, 255, 256, FEAT);

  // --- test branch: conv_bn_relu x2 on feat2 -> c3_t ---
  k_nchw_to_nhwc<<<blocks(FEAT), 256, 0, stream>>>(feat2, bufA, FEAT);
  k_conv3x3_wmma<<<convGrid, 256, 0, stream>>>(bufA, w1t, b_c1t, ybuf);
  k_bn_stats<<<324, 256, 0, stream>>>(ybuf, part, 82944, 256, 256);
  k_bn_finalize<<<1, 256, 0, stream>>>(part, g_c1t, be_c1t, ab, 324, 256, 82944.f);
  k_bn_apply<<<blocks(FEAT), 256, 0, stream>>>(ybuf, ab, bufB, nullptr, 255, 256, FEAT);

  k_conv3x3_wmma<<<convGrid, 256, 0, stream>>>(bufB, w2t, b_c2t, ybuf);
  k_bn_stats<<<324, 256, 0, stream>>>(ybuf, part, 82944, 256, 256);
  k_bn_finalize<<<1, 256, 0, stream>>>(part, g_c2t, be_c2t, ab, 324, 256, 82944.f);
  k_bn_apply<<<blocks(FEAT), 256, 0, stream>>>(ybuf, ab, bufA, nullptr, 255, 256, FEAT); // c3_t

  // --- modulation: PrRoIPool(5x5) -> f1r -> f2r ---
  k_pool_r<<<64, 256, 0, stream>>>(bufC, bb1, poolr);

  k_gemm_wmma<<<dim3(16, 1), 128, 0, stream>>>(poolr, wf1r, b_f1r, y1, 1024, 6400);
  k_bn_stats<<<1, 256, 0, stream>>>(y1, part, 64, 1024, 64);
  k_bn_finalize<<<1, 256, 0, stream>>>(part, g_f1r, be_f1r, ab, 1, 1024, 64.f);
  k_bn_apply<<<blocks(65536), 256, 0, stream>>>(y1, ab, m_h, nullptr, 1023, 1024, 65536L);

  k_gemm_wmma<<<dim3(4, 1), 128, 0, stream>>>(m_h, wf2r, b_f2r, y2, 256, 1024);
  k_bn_stats<<<1, 256, 0, stream>>>(y2, part, 64, 256, 64);
  k_bn_finalize<<<1, 256, 0, stream>>>(part, g_f2r, be_f2r, ab, 1, 256, 64.f);
  k_bn_apply<<<blocks(16384), 256, 0, stream>>>(y2, ab, modh, modf, 255, 256, 16384L);

  // --- predict_iou: modulated PrRoIPool(7x7) -> f1rt -> f2rt -> iou ---
  k_pool_t<<<1024, 256, 0, stream>>>(bufA, props, modf, poolt);

  k_gemm128_wmma<<<dim3(8, 8), 256, 0, stream>>>(poolt, wf1rt, b_f1rt, z1, 1024, 12544);
  k_bn_stats<<<32, 256, 0, stream>>>(z1, part, 1024, 1024, 32);
  k_bn_finalize<<<1, 256, 0, stream>>>(part, g_f1rt, be_f1rt, ab, 32, 1024, 1024.f);
  k_bn_apply<<<blocks(1048576), 256, 0, stream>>>(z1, ab, z1h, nullptr, 1023, 1024, 1048576L);

  k_gemm128_wmma<<<dim3(8, 8), 256, 0, stream>>>(z1h, wf2rt, b_f2rt, z2, 1024, 1024);
  k_bn_stats<<<32, 256, 0, stream>>>(z2, part, 1024, 1024, 32);
  k_bn_finalize<<<1, 256, 0, stream>>>(part, g_f2rt, be_f2rt, ab, 32, 1024, 1024.f);
  k_bn_apply<<<blocks(1048576), 256, 0, stream>>>(z2, ab, z2h, nullptr, 1023, 1024, 1048576L);

  k_iou<<<1024, 256, 0, stream>>>(z2h, w_iou, b_iou, out);
}
